// BidirectionalAttention_38044820308306
// MI455X (gfx1250) — matrix operand order repaired and checked by reading, not compiled, not run on hardware
//
#include <hip/hip_runtime.h>
#include <hip/hip_bf16.h>

// ---------------------------------------------------------------------------
// Bidirectional co-attention, CDNA5 (gfx1250), wave32 + WMMA bf16 pipeline.
// B=16, T1=T2=1024, K1D=K2D=512, V1D=V2D=256, A=128.
// N-blocked WMMA tiles: each wave owns 16xN (N=32/64) so the A-fragment is
// reused across 2-4 v_wmma issues per K-step.
// ---------------------------------------------------------------------------

typedef __bf16 bf16;
typedef __attribute__((ext_vector_type(16))) bf16  v16bf;
typedef __attribute__((ext_vector_type(8)))  bf16  v8bf;
typedef __attribute__((ext_vector_type(8)))  float v8f;

#define DEVINL __device__ __forceinline__

// ---------------- problem constants ----------------
namespace cfg {
constexpr int Bb  = 16;
constexpr int T1  = 1024;
constexpr int T2  = 1024;
constexpr int K1D = 512;
constexpr int K2D = 512;
constexpr int V1D = 256;
constexpr int V2D = 256;
constexpr int A   = 128;
}

// ---------------- small converters ----------------
__global__ __launch_bounds__(256) void cvtw_kernel(const float* __restrict__ W1,
                                                   const float* __restrict__ W2,
                                                   bf16* __restrict__ W1b,
                                                   bf16* __restrict__ W2b, int n) {
    int i = blockIdx.x * blockDim.x + threadIdx.x;
    if (i < n) {
        W1b[i] = (bf16)W1[i];
        W2b[i] = (bf16)W2[i];
    }
}

// V [B,T,N] f32  ->  Vt [B,N,T] bf16   (32x32 LDS tiles, padded vs bank conflicts)
__global__ __launch_bounds__(256) void vt_kernel(const float* __restrict__ V,
                                                 bf16* __restrict__ Vt,
                                                 int Tv, int Nv) {
    __shared__ float tile[32][33];
    const int b  = blockIdx.z;
    const int t0 = blockIdx.x * 32;
    const int n0 = blockIdx.y * 32;
    const int tx = threadIdx.x & 31;
    const int ty = threadIdx.x >> 5;   // 8 rows per pass
    #pragma unroll
    for (int r = ty; r < 32; r += 8)
        tile[r][tx] = V[((size_t)b * Tv + t0 + r) * Nv + n0 + tx];
    __syncthreads();
    #pragma unroll
    for (int r = ty; r < 32; r += 8)
        Vt[((size_t)b * Nv + n0 + r) * Tv + t0 + tx] = (bf16)tile[tx][r];
}

// ---------------- WMMA fragment helpers (layouts per ISA 7.12.2) ----------------
// A fragment (16x32 bf16) from a row-major bf16 row pointer.
DEVINL v16bf load_a_bf16(const bf16* __restrict__ row, int kk, bool hi) {
    const bf16* ap = row + kk + (hi ? 8 : 0);
    v8bf a0 = *(const v8bf*)(ap);
    v8bf a1 = *(const v8bf*)(ap + 16);
    v16bf a;
    #pragma unroll
    for (int t = 0; t < 8; ++t) { a[t] = a0[t]; a[t + 8] = a1[t]; }
    return a;
}
// A fragment converted on-the-fly from f32 source.
DEVINL v16bf load_a_f32(const float* __restrict__ row, int kk, bool hi) {
    const float* ap = row + kk + (hi ? 8 : 0);
    v8f x0 = *(const v8f*)(ap);
    v8f x1 = *(const v8f*)(ap + 16);
    v16bf a;
    #pragma unroll
    for (int t = 0; t < 8; ++t) { a[t] = (bf16)x0[t]; a[t + 8] = (bf16)x1[t]; }
    return a;
}
// B fragment (32x16 bf16): column n of B == row n of K-contiguous storage.
DEVINL v16bf load_b_bf16(const bf16* __restrict__ row, int kk, bool hi) {
    const bf16* bp = row + kk + (hi ? 8 : 0);
    v8bf b0 = *(const v8bf*)(bp);
    v8bf b1 = *(const v8bf*)(bp + 16);
    v16bf b;
    #pragma unroll
    for (int t = 0; t < 8; ++t) { b[t] = b0[t]; b[t + 8] = b1[t]; }
    return b;
}

// ---------------- projection: Q = X @ W^T + b  (bf16 out) ----------------
// Wave tile: 16 (M) x 32 (N). Block = 8 waves: 2 M-tiles x (4 waves covering A=128).
__global__ __launch_bounds__(256) void proj_kernel(const float* __restrict__ X,   // [M,KD] f32
                                                   const bf16* __restrict__ Wb,   // [A,KD] bf16
                                                   const float* __restrict__ bias,// [A]
                                                   bf16* __restrict__ Q,          // [M,A] bf16
                                                   int M, int KD, int Aout) {
    const int wave = threadIdx.x >> 5;
    const int lane = threadIdx.x & 31;
    const int m0 = blockIdx.x * 32 + (wave >> 2) * 16;
    const int n0 = (wave & 3) * 32;
    const int lr = lane & 15;
    const bool hi = lane >= 16;

    const float* xrow = X + (size_t)(m0 + lr) * KD;
    const bf16* wrow0 = Wb + (size_t)(n0 + lr) * KD;
    const bf16* wrow1 = Wb + (size_t)(n0 + 16 + lr) * KD;

    v8f c0 = {}, c1 = {};
    for (int kk = 0; kk < KD; kk += 32) {
        __builtin_prefetch(xrow + kk + 128, 0, 1);
        v16bf a = load_a_f32(xrow, kk, hi);
        v16bf b0 = load_b_bf16(wrow0, kk, hi);
        v16bf b1 = load_b_bf16(wrow1, kk, hi);
        c0 = __builtin_amdgcn_wmma_f32_16x16x32_bf16(false, a, false, b0, (short)0, c0,
                                                     false, false);
        c1 = __builtin_amdgcn_wmma_f32_16x16x32_bf16(false, a, false, b1, (short)0, c1,
                                                     false, false);
    }
    const float bv0 = bias[n0 + lr];
    const float bv1 = bias[n0 + 16 + lr];
    #pragma unroll
    for (int r = 0; r < 8; ++r) {
        int m = m0 + r + (hi ? 8 : 0);
        Q[(size_t)m * Aout + n0 + lr]      = (bf16)(c0[r] + bv0);
        Q[(size_t)m * Aout + n0 + 16 + lr] = (bf16)(c1[r] + bv1);
    }
}

// ---------------- score: S = Q1 @ Q2^T, XOR pad mask, f32 out ----------------
// Wave tile: 16 (i) x 64 (j): A-fragment reused across 4 WMMAs per K-step.
__global__ __launch_bounds__(256) void score_kernel(const bf16* __restrict__ Q1,  // [B*T1,A]
                                                    const bf16* __restrict__ Q2,  // [B*T2,A]
                                                    const int* __restrict__ len1,
                                                    const int* __restrict__ len2,
                                                    float* __restrict__ S,        // [B,T1,T2]
                                                    int T1v, int T2v, int Ain) {
    const int wave = threadIdx.x >> 5;
    const int lane = threadIdx.x & 31;
    const int b  = blockIdx.z;
    const int i0 = blockIdx.y * 16;
    const int j0 = (blockIdx.x * 8 + wave) * 64;
    const int lr = lane & 15;
    const bool hi = lane >= 16;

    const bf16* q1row = Q1 + ((size_t)b * T1v + i0 + lr) * Ain;
    const bf16* q2base = Q2 + (size_t)b * T2v * Ain;
    const bf16* q2row[4];
    #pragma unroll
    for (int t = 0; t < 4; ++t)
        q2row[t] = q2base + (size_t)(j0 + t * 16 + lr) * Ain;

    v8f c[4] = {};
    #pragma unroll
    for (int kk = 0; kk < 128; kk += 32) {
        v16bf a = load_a_bf16(q1row, kk, hi);
        #pragma unroll
        for (int t = 0; t < 4; ++t) {
            v16bf bb = load_b_bf16(q2row[t], kk, hi);
            c[t] = __builtin_amdgcn_wmma_f32_16x16x32_bf16(false, a, false, bb, (short)0,
                                                           c[t], false, false);
        }
    }
    const int L1 = len1[b], L2 = len2[b];
    float* srow = S + (size_t)b * T1v * T2v;
    const float ninf = -__builtin_inff();
    #pragma unroll
    for (int t = 0; t < 4; ++t) {
        const int j = j0 + t * 16 + lr;
        const bool cp = j >= L2;
        #pragma unroll
        for (int r = 0; r < 8; ++r) {
            int i = i0 + r + (hi ? 8 : 0);
            bool rp = i >= L1;
            srow[(size_t)i * T2v + j] = (rp != cp) ? ninf : c[t][r];
        }
    }
}

// ---------------- row softmax: w2 = softmax(score, axis=-1) ----------------
__global__ __launch_bounds__(256) void softmax_row_kernel(const float* __restrict__ S,
                                                          float* __restrict__ Wf,
                                                          bf16* __restrict__ Wb,
                                                          int T2v) {
    __shared__ float red[8];
    const size_t row = blockIdx.x;
    const float* s = S + row * T2v;
    const int tid = threadIdx.x, lane = tid & 31, wave = tid >> 5;

    float vals[4];
    float m = -__builtin_inff();
    #pragma unroll
    for (int t = 0; t < 4; ++t) { vals[t] = s[tid + t * 256]; m = fmaxf(m, vals[t]); }
    #pragma unroll
    for (int off = 16; off; off >>= 1) m = fmaxf(m, __shfl_xor(m, off, 32));
    if (lane == 0) red[wave] = m;
    __syncthreads();
    float x = red[lane & 7];
    #pragma unroll
    for (int off = 4; off; off >>= 1) x = fmaxf(x, __shfl_xor(x, off, 32));
    m = x;

    float sum = 0.f;
    #pragma unroll
    for (int t = 0; t < 4; ++t) { vals[t] = __expf(vals[t] - m); sum += vals[t]; }
    #pragma unroll
    for (int off = 16; off; off >>= 1) sum += __shfl_xor(sum, off, 32);
    __syncthreads();
    if (lane == 0) red[wave] = sum;
    __syncthreads();
    float xs = red[lane & 7];
    #pragma unroll
    for (int off = 4; off; off >>= 1) xs += __shfl_xor(xs, off, 32);
    const float inv = 1.0f / xs;

    float* wf = Wf + row * T2v;
    bf16*  wb = Wb + row * T2v;
    #pragma unroll
    for (int t = 0; t < 4; ++t) {
        float w = vals[t] * inv;
        wf[tid + t * 256] = w;
        wb[tid + t * 256] = (bf16)w;
    }
}

// ---------------- column softmax: w1[b,j,i] = softmax_i(score[b,i,j]) ----------------
__global__ __launch_bounds__(256) void softmax_col_kernel(const float* __restrict__ S,  // [B,T1,T2]
                                                          float* __restrict__ W1f,      // [B,T2,T1]
                                                          bf16* __restrict__ W1b,
                                                          int T1v, int T2v) {
    __shared__ float red[16][17];
    const int b  = blockIdx.y;
    const int tj = threadIdx.x;                 // 16 columns, coalesced reads
    const int ti = threadIdx.y;                 // 16 row-partials
    const int j  = blockIdx.x * 16 + tj;
    const float* s = S + (size_t)b * T1v * T2v + j;

    float m = -__builtin_inff();
    for (int ii = ti; ii < T1v; ii += 16) m = fmaxf(m, s[(size_t)ii * T2v]);
    red[ti][tj] = m;
    __syncthreads();
    #pragma unroll
    for (int off = 8; off; off >>= 1) {
        if (ti < off) red[ti][tj] = fmaxf(red[ti][tj], red[ti + off][tj]);
        __syncthreads();
    }
    m = red[0][tj];
    __syncthreads();

    float sum = 0.f;
    for (int ii = ti; ii < T1v; ii += 16) sum += __expf(s[(size_t)ii * T2v] - m);
    red[ti][tj] = sum;
    __syncthreads();
    #pragma unroll
    for (int off = 8; off; off >>= 1) {
        if (ti < off) red[ti][tj] += red[ti + off][tj];
        __syncthreads();
    }
    const float inv = 1.0f / red[0][tj];

    float* wrow  = W1f + ((size_t)b * T2v + j) * T1v;
    bf16*  wrowb = W1b + ((size_t)b * T2v + j) * T1v;
    for (int ii = ti; ii < T1v; ii += 16) {
        float w = __expf(s[(size_t)ii * T2v] - m) * inv;
        wrow[ii]  = w;
        wrowb[ii] = (bf16)w;
    }
}

// ---------------- output GEMM: O = W @ V   (A=weights bf16, B=v^T bf16) ----------------
// Wave tile: 16 (M) x 64 (N). Block = 8 waves: 2 M-tiles x (4 waves covering N=256).
__global__ __launch_bounds__(256) void av_kernel(const bf16* __restrict__ Wm,  // [B,M,K] bf16
                                                 const bf16* __restrict__ Vt,  // [B,N,K] bf16
                                                 float* __restrict__ O,        // [B,M,N] f32
                                                 int Mv, int Nv, int Kv) {
    const int wave = threadIdx.x >> 5;
    const int lane = threadIdx.x & 31;
    const int b  = blockIdx.z;
    const int m0 = blockIdx.y * 32 + (wave >> 2) * 16;
    const int n0 = (wave & 3) * 64;            // Nv = 256
    const int lr = lane & 15;
    const bool hi = lane >= 16;

    const bf16* arow = Wm + ((size_t)b * Mv + m0 + lr) * Kv;
    const bf16* vbase = Vt + (size_t)b * Nv * Kv;
    const bf16* brow[4];
    #pragma unroll
    for (int t = 0; t < 4; ++t)
        brow[t] = vbase + (size_t)(n0 + t * 16 + lr) * Kv;

    v8f c[4] = {};
    #pragma unroll 2
    for (int kk = 0; kk < Kv; kk += 32) {
        __builtin_prefetch(arow + kk + 256, 0, 1);
        v16bf a = load_a_bf16(arow, kk, hi);
        #pragma unroll
        for (int t = 0; t < 4; ++t) {
            v16bf bb = load_b_bf16(brow[t], kk, hi);
            c[t] = __builtin_amdgcn_wmma_f32_16x16x32_bf16(false, a, false, bb, (short)0,
                                                           c[t], false, false);
        }
    }
    float* orow = O + (size_t)b * Mv * Nv;
    #pragma unroll
    for (int t = 0; t < 4; ++t) {
        #pragma unroll
        for (int r = 0; r < 8; ++r) {
            int m = m0 + r + (hi ? 8 : 0);
            orow[(size_t)m * Nv + n0 + t * 16 + lr] = c[t][r];
        }
    }
}

// ---------------------------------------------------------------------------
extern "C" void kernel_launch(void* const* d_in, const int* in_sizes, int n_in,
                              void* d_out, int out_size, void* d_ws, size_t ws_size,
                              hipStream_t stream) {
    using namespace cfg;
    const float* k1 = (const float*)d_in[0];
    const float* k2 = (const float*)d_in[1];
    const float* v1 = (const float*)d_in[2];
    const float* v2 = (const float*)d_in[3];
    const float* W1 = (const float*)d_in[4];
    const float* b1 = (const float*)d_in[5];
    const float* W2 = (const float*)d_in[6];
    const float* b2 = (const float*)d_in[7];
    const int*   l1 = (const int*)d_in[8];
    const int*   l2 = (const int*)d_in[9];
    float* out = (float*)d_out;

    // d_out layout (flat f32, reference return order): o1, o2, w1, w2, score
    const size_t OFF_O1 = 0;
    const size_t OFF_O2 = OFF_O1 + (size_t)Bb * T2 * V1D;
    const size_t OFF_W1 = OFF_O2 + (size_t)Bb * T1 * V2D;
    const size_t OFF_W2 = OFF_W1 + (size_t)Bb * T2 * T1;
    const size_t OFF_S  = OFF_W2 + (size_t)Bb * T1 * T2;

    // workspace layout (bytes)
    char* ws = (char*)d_ws;
    size_t o = 0;
    bf16* q1b = (bf16*)(ws + o); o += (size_t)Bb * T1 * A * sizeof(bf16);
    bf16* q2b = (bf16*)(ws + o); o += (size_t)Bb * T2 * A * sizeof(bf16);
    bf16* v1t = (bf16*)(ws + o); o += (size_t)Bb * V1D * T1 * sizeof(bf16);
    bf16* v2t = (bf16*)(ws + o); o += (size_t)Bb * V2D * T2 * sizeof(bf16);
    bf16* w1b = (bf16*)(ws + o); o += (size_t)Bb * T2 * T1 * sizeof(bf16);
    bf16* w2b = (bf16*)(ws + o); o += (size_t)Bb * T1 * T2 * sizeof(bf16);
    bf16* W1b = (bf16*)(ws + o); o += (size_t)A * K1D * sizeof(bf16);
    bf16* W2b = (bf16*)(ws + o); o += (size_t)A * K2D * sizeof(bf16);

    // 1) convert weights, transpose values
    {
        int n = A * K1D;
        cvtw_kernel<<<(n + 255) / 256, 256, 0, stream>>>(W1, W2, W1b, W2b, n);
    }
    vt_kernel<<<dim3(T1 / 32, V1D / 32, Bb), 256, 0, stream>>>(v1, v1t, T1, V1D);
    vt_kernel<<<dim3(T2 / 32, V2D / 32, Bb), 256, 0, stream>>>(v2, v2t, T2, V2D);

    // 2) projections q1,q2 (WMMA bf16, 16x32 wave tiles)
    proj_kernel<<<Bb * T1 / 32, 256, 0, stream>>>(k1, W1b, b1, q1b, Bb * T1, K1D, A);
    proj_kernel<<<Bb * T2 / 32, 256, 0, stream>>>(k2, W2b, b2, q2b, Bb * T2, K2D, A);

    // 3) masked score (WMMA bf16, 16x64 wave tiles, f32 straight into d_out)
    score_kernel<<<dim3(T2 / 512, T1 / 16, Bb), 256, 0, stream>>>(
        q1b, q2b, l1, l2, out + OFF_S, T1, T2, A);

    // 4) dual softmaxes
    softmax_row_kernel<<<Bb * T1, 256, 0, stream>>>(out + OFF_S, out + OFF_W2, w2b, T2);
    softmax_col_kernel<<<dim3(T2 / 16, Bb), dim3(16, 16), 0, stream>>>(
        out + OFF_S, out + OFF_W1, w1b, T1, T2);

    // 5) outputs: o1 = w1 @ v1, o2 = w2 @ v2 (WMMA bf16, 16x64 wave tiles)
    av_kernel<<<dim3(1, T2 / 32, Bb), 256, 0, stream>>>(
        w1b, v1t, out + OFF_O1, T2, V1D, T1);
    av_kernel<<<dim3(1, T1 / 32, Bb), 256, 0, stream>>>(
        w2b, v2t, out + OFF_O2, T1, V2D, T2);
}
